// InstanceDeformableConv_30846455120524
// MI455X (gfx1250) — compile-verified, matrix operand order checked by reading, not run on hardware
//
#include <hip/hip_runtime.h>

// ---------------------------------------------------------------------------
// MI455X (gfx1250) implementation: bf16 WMMA implicit-GEMM convs + fused
// deformable-conv pipeline. fp32 in/out, bf16 WMMA (f32 accum) for all GEMMs.
// Double-buffered LDS staging; async global->LDS DMA when the toolchain
// exposes the gfx1250 async builtins.
// ---------------------------------------------------------------------------

typedef __attribute__((ext_vector_type(16))) __bf16 v16bf;
typedef __attribute__((ext_vector_type(8)))  float  v8f;

union FragAB { v16bf v; unsigned u[8]; };
union FragC  { v8f v; float f[8]; };

#if defined(__has_builtin)
#  if __has_builtin(__builtin_amdgcn_global_load_async_to_lds_b128)
#    define HAVE_ASYNC_LDS 1
#  endif
#endif
#ifndef HAVE_ASYNC_LDS
#  define HAVE_ASYNC_LDS 0
#endif

#if HAVE_ASYNC_LDS
typedef __attribute__((__vector_size__(16))) int i32x4;
typedef __attribute__((address_space(1))) i32x4 gl_i32x4;   // global (AS1)
typedef __attribute__((address_space(3))) i32x4 ld_i32x4;   // LDS (AS3)
#endif

__device__ __forceinline__ void wait_async0() {
#if defined(__has_builtin) && __has_builtin(__builtin_amdgcn_s_wait_asynccnt)
  __builtin_amdgcn_s_wait_asynccnt(0);
#else
  asm volatile("s_wait_asynccnt 0x0" ::: "memory");
#endif
}

__device__ __forceinline__ unsigned short f2bf(float f) {
  union { float f; unsigned u; } t; t.f = f;
  unsigned u = t.u;
  unsigned r = 0x7FFFu + ((u >> 16) & 1u);   // round-to-nearest-even
  return (unsigned short)((u + r) >> 16);
}
__device__ __forceinline__ unsigned packbf(float lo, float hi) {
  return (unsigned)f2bf(lo) | ((unsigned)f2bf(hi) << 16);
}

// ---------------------------------------------------------------------------
// Implicit-GEMM conv via V_WMMA_F32_16X16X32_BF16.
//   MODE 0: input X fp32 NCHW, weights flat [Cout][Cin*KHW] (OIHW order),
//           K index = c*KHW + ky*KW + kx (KHW = 9 or 1, compile-time).
//   MODE 1: input Bim bf16 im2col [P][Ktot] (Ktot=1152, K = kk*128 + c),
//           weights w_def [Cout][128][9] -> element (co,K) at c*9+kk.
// Block: 256 threads = 8 waves; block owns 16 output positions, wave w owns
// rows [16w,16w+16). 32x16 bf16 B panel double-buffered through LDS
// (async global->LDS b128 DMA in MODE 1 when available).
// Epilogue fuses conv-bias + optional BN + optional ReLU -> fp32 NCHW.
// ---------------------------------------------------------------------------
template <int MODE, int KHW>
__global__ __launch_bounds__(256)
void gemm_conv(const float* __restrict__ X, const unsigned short* __restrict__ Bim,
               const float* __restrict__ Wt, const float* __restrict__ bias,
               const float* __restrict__ gam, const float* __restrict__ bet,
               const float* __restrict__ mu,  const float* __restrict__ var,
               float* __restrict__ Y,
               int Nn, int Cin, int Cout, int H, int W, int Ktot, int relu)
{
  constexpr int PAD = (KHW == 9) ? 1 : 0;
  __shared__ __align__(16) unsigned ldsB[2][256];   // [buf][16 cols][16 K-pairs]

  const int tid  = threadIdx.x;
  const int lane = tid & 31;
  const int wave = tid >> 5;
  const int HW = H * W;
  const int P = Nn * HW;
  const int colBase = blockIdx.x * 16;
  const int coBase  = wave * 16;

  // staging coordinates: thread t loads K-pair (t&15) of column (t>>4)
  const int s_n  = tid >> 4;
  const int s_kp = tid & 15;
  const int sp = colBase + s_n;
  const bool spv = sp < P;
  int sy = 0, sx = 0;
  const float* Xn = X;
  if (MODE == 0 && spv) {
    const int sn = sp / HW; const int r = sp - sn * HW;
    sy = r / W; sx = r - sy * W;
    Xn = X + (size_t)sn * Cin * HW;          // per-image base
  }

  const int grp  = lane >> 4;   // lane group (K-half of A/B fragments)
  const int mrow = lane & 15;   // A: row M;  B: column N
  int coc = coBase + mrow; if (coc >= Cout) coc = Cout - 1;

  FragC acc;
  #pragma unroll
  for (int j = 0; j < 8; ++j) acc.f[j] = 0.f;

  // ---- B-panel staging (one 32x16 bf16 panel -> ldsB[bi]) ----
  auto stage = [&](int kbase, int bi) {
    if (MODE == 0) {
      unsigned pk = 0;
      if (spv) {
        const int k0 = kbase + (s_kp << 1);
        float f0 = 0.f, f1 = 0.f;
        {
          const int k = k0;
          const int c = k / KHW; const int rr = k - c * KHW;
          const int ky = (KHW == 9) ? (rr / 3) : 0;
          const int kx = (KHW == 9) ? (rr - ky * 3) : 0;
          const int iy = sy - PAD + ky, ix = sx - PAD + kx;
          if (iy >= 0 && iy < H && ix >= 0 && ix < W)
            f0 = Xn[(c * H + iy) * W + ix];
        }
        {
          const int k = k0 + 1;
          const int c = k / KHW; const int rr = k - c * KHW;
          const int ky = (KHW == 9) ? (rr / 3) : 0;
          const int kx = (KHW == 9) ? (rr - ky * 3) : 0;
          const int iy = sy - PAD + ky, ix = sx - PAD + kx;
          if (iy >= 0 && iy < H && ix >= 0 && ix < W)
            f1 = Xn[(c * H + iy) * W + ix];
        }
        pk = packbf(f0, f1);
      }
      ldsB[bi][(s_n << 4) | s_kp] = pk;
    } else {
#if HAVE_ASYNC_LDS
      // 64 threads x b128 async DMA: global (bf16 [P][Ktot]) -> LDS panel.
      if (tid < 64) {
        const int col = tid >> 2, q = tid & 3;
        const int csp = colBase + col;
        if (csp < P) {
          const unsigned short* g = Bim + (size_t)csp * Ktot + kbase + (q << 3);
          __builtin_amdgcn_global_load_async_to_lds_b128(
              (gl_i32x4*)g,
              (ld_i32x4*)&ldsB[bi][(col << 4) + (q << 2)],
              0, 0);
        } else {
          uint4 z; z.x = z.y = z.z = z.w = 0u;
          *(uint4*)&ldsB[bi][(col << 4) + (q << 2)] = z;
        }
      }
#else
      unsigned pk = 0;
      if (spv)
        pk = *(const unsigned*)(Bim + (size_t)sp * Ktot + kbase + (s_kp << 1));
      ldsB[bi][(s_n << 4) | s_kp] = pk;
#endif
    }
  };

  stage(0, 0);
  if (MODE == 1 && HAVE_ASYNC_LDS) wait_async0();
  __syncthreads();

  int cur = 0;
  for (int kbase = 0; kbase < Ktot; kbase += 32) {
    // ---- B fragment from LDS (16-bit B layout: lanes=N, grp picks K half) ----
    FragAB bfr;
    {
      const unsigned* base = &ldsB[cur][(mrow << 4) + (grp << 2)];
      *(uint4*)&bfr.u[0] = *(const uint4*)&base[0];   // V0..V3
      *(uint4*)&bfr.u[4] = *(const uint4*)&base[8];   // V4..V7
    }

    // ---- prefetch + stage next panel into the other buffer ----
    const int knext = kbase + 32;
    if (MODE == 0 && knext < Ktot)
      __builtin_prefetch(Wt + (size_t)coc * Ktot + knext, 0, 0);
    if (knext < Ktot) stage(knext, cur ^ 1);

    // ---- A fragment from global (16-bit A layout) ----
    FragAB afr;
    #pragma unroll
    for (int v2 = 0; v2 < 8; ++v2) {
      const int kb = ((v2 & 4) << 2) + (grp << 3) + ((v2 & 3) << 1);
      const int K = kbase + kb;
      if (MODE == 0) {
        const float2 wp = *(const float2*)(Wt + (size_t)coc * Ktot + K);
        afr.u[v2] = packbf(wp.x, wp.y);
      } else {
        const int kk0 = K >> 7,       c0 = K & 127;
        const int kk1 = (K + 1) >> 7, c1 = (K + 1) & 127;
        const float w0 = Wt[(size_t)coc * Ktot + c0 * 9 + kk0];
        const float w1 = Wt[(size_t)coc * Ktot + c1 * 9 + kk1];
        afr.u[v2] = packbf(w0, w1);
      }
    }

    acc.v = __builtin_amdgcn_wmma_f32_16x16x32_bf16(
        false, afr.v, false, bfr.v, (short)0, acc.v, false, false);

    if (MODE == 1 && HAVE_ASYNC_LDS) wait_async0();
    __syncthreads();
    cur ^= 1;
  }

  // ---- epilogue: bias + optional BN + optional ReLU, scatter to NCHW ----
  if (coBase >= Cout) return;
  const int pcol = colBase + mrow;
  if (pcol >= P) return;
  const int n = pcol / HW;
  const int r = pcol - n * HW;
  const int y = r / W, x = r - (r / W) * W;
  float* Yn = Y + (size_t)n * Cout * HW;
  #pragma unroll
  for (int rr = 0; rr < 8; ++rr) {
    const int co = coBase + rr + (grp << 3);   // D layout: VGPR rr -> M=rr / 8+rr
    if (co >= Cout) continue;
    float val = acc.f[rr] + bias[co];
    if (gam) {
      const float s = gam[co] * rsqrtf(var[co] + 1e-5f);
      val = val * s + (bet[co] - mu[co] * s);
    }
    if (relu) val = fmaxf(val, 0.f);
    Yn[(co * H + y) * W + x] = val;
  }
}

// ---------------------------------------------------------------------------
// NCHW -> NHWC transpose (fp32) for channel-contiguous bilinear gathers.
// ---------------------------------------------------------------------------
__global__ void nchw_to_nhwc(const float* __restrict__ in, float* __restrict__ out,
                             int Nn, int C, int H, int W)
{
  size_t idx = (size_t)blockIdx.x * blockDim.x + threadIdx.x;
  const size_t total = (size_t)Nn * C * H * W;
  if (idx >= total) return;
  const int x = idx % W; size_t t = idx / W;
  const int y = t % H; t /= H;
  const int c = t % C; const int n = (int)(t / C);
  out[(((size_t)n * H + y) * W + x) * C + c] = in[idx];
}

// ---------------------------------------------------------------------------
// Modulated-deformable im2col: bilinear sample ht (NHWC fp32) at offset
// positions, scale by 2*sigmoid(mask), write bf16 im2col [P][1152] with
// K = k*128 + c  (coalesced stores across the 128 c-lanes).
// ---------------------------------------------------------------------------
__global__ __launch_bounds__(128)
void deform_im2col(const float* __restrict__ ht, const float* __restrict__ off,
                   const float* __restrict__ msk, unsigned short* __restrict__ Vbuf,
                   int Nn, int H, int W)
{
  const int C = 128;
  const int c = threadIdx.x;
  const int hw = blockIdx.x;
  const int n  = blockIdx.y;
  const int y = hw / W, x = hw - y * W;
  const size_t HW = (size_t)H * W;
  const size_t p = (size_t)n * HW + hw;

  #pragma unroll
  for (int k = 0; k < 9; ++k) {
    const float dy = off[(size_t)(n * 18 + 2 * k)     * HW + hw];
    const float dx = off[(size_t)(n * 18 + 2 * k + 1) * HW + hw];
    const float mr = msk[(size_t)(n * 9 + k) * HW + hw];
    const float mv = 2.f / (1.f + __expf(-mr));
    const int ky = k / 3, kx = k - ky * 3;
    const float py = (float)y - 1.f + (float)ky + dy;
    const float px = (float)x - 1.f + (float)kx + dx;
    const float y0f = floorf(py), x0f = floorf(px);
    const float wy = py - y0f, wx = px - x0f;
    const int y0 = (int)y0f, x0 = (int)x0f;
    float v = 0.f;
    #pragma unroll
    for (int dyi = 0; dyi < 2; ++dyi) {
      #pragma unroll
      for (int dxi = 0; dxi < 2; ++dxi) {
        const int yy = y0 + dyi, xx = x0 + dxi;
        const float wgt = (dyi ? wy : 1.f - wy) * (dxi ? wx : 1.f - wx);
        if (yy >= 0 && yy < H && xx >= 0 && xx < W)
          v += wgt * ht[((size_t)(n * H + yy) * W + xx) * C + c];
      }
    }
    Vbuf[p * 1152 + k * C + c] = f2bf(v * mv);
  }
}

// ---------------------------------------------------------------------------
// relu(d + res) followed by stride-1 2x2 maxpool with pad 1 -> (H+1)x(W+1).
// relu output >= 0 and every window has >=1 valid cell, so init best = 0.
// ---------------------------------------------------------------------------
__global__ void add_relu_pool(const float* __restrict__ d, const float* __restrict__ res,
                              float* __restrict__ out, int Nn, int C, int H, int W)
{
  const int Ho = H + 1, Wo = W + 1;
  size_t idx = (size_t)blockIdx.x * blockDim.x + threadIdx.x;
  const size_t total = (size_t)Nn * C * Ho * Wo;
  if (idx >= total) return;
  const int xo = idx % Wo; size_t t = idx / Wo;
  const int yo = t % Ho; t /= Ho;
  const int c = t % C; const int n = (int)(t / C);
  float best = 0.f;
  #pragma unroll
  for (int dyi = 0; dyi < 2; ++dyi)
    #pragma unroll
    for (int dxi = 0; dxi < 2; ++dxi) {
      const int yy = yo - 1 + dyi, xx = xo - 1 + dxi;
      if (yy >= 0 && yy < H && xx >= 0 && xx < W) {
        const size_t o = ((size_t)(n * C + c) * H + yy) * W + xx;
        best = fmaxf(best, d[o] + res[o]);
      }
    }
  out[idx] = best;
}

// ---------------------------------------------------------------------------
extern "C" void kernel_launch(void* const* d_in, const int* in_sizes, int n_in,
                              void* d_out, int out_size, void* d_ws, size_t ws_size,
                              hipStream_t stream)
{
  const float* x    = (const float*)d_in[0];
  const float* w1   = (const float*)d_in[1];
  const float* b1   = (const float*)d_in[2];
  const float* g1   = (const float*)d_in[3];
  const float* be1  = (const float*)d_in[4];
  const float* m1   = (const float*)d_in[5];
  const float* v1   = (const float*)d_in[6];
  const float* w_off= (const float*)d_in[7];
  const float* b_off= (const float*)d_in[8];
  const float* w_mod= (const float*)d_in[9];
  const float* b_mod= (const float*)d_in[10];
  const float* w_def= (const float*)d_in[11];
  const float* b_def= (const float*)d_in[12];
  const float* g2   = (const float*)d_in[13];
  const float* be2  = (const float*)d_in[14];
  const float* m2   = (const float*)d_in[15];
  const float* v2   = (const float*)d_in[16];
  const float* w_ds = (const float*)d_in[17];
  const float* b_ds = (const float*)d_in[18];
  const float* g3   = (const float*)d_in[19];
  const float* be3  = (const float*)d_in[20];
  const float* m3   = (const float*)d_in[21];
  const float* v3   = (const float*)d_in[22];
  (void)in_sizes; (void)n_in; (void)out_size; (void)ws_size;

  char* wsp = (char*)d_ws;
  auto carve = [&](size_t bytes) -> char* {
    char* pp = wsp; wsp += (bytes + 255) & ~(size_t)255; return pp;
  };
  const size_t SZ = (size_t)4 * 128 * 66 * 66;          // max NCHW fp32 buffer
  float* hb   = (float*)carve(SZ * 4);                  // conv1 output (h)
  float* htb  = (float*)carve(SZ * 4);                  // h in NHWC
  float* offb = (float*)carve((size_t)4 * 18 * 66 * 66 * 4);
  float* mskb = (float*)carve((size_t)4 * 9  * 66 * 66 * 4);
  unsigned short* vb = (unsigned short*)carve((size_t)4 * 66 * 66 * 1152 * 2);
  float* db   = (float*)carve(SZ * 4);                  // deform conv + bn
  float* resb = (float*)carve(SZ * 4);                  // downsample + bn
  float* xb   = (float*)carve(SZ * 4);                  // inter-block activation

  const float* xin = x;
  for (int i = 0; i < 2; ++i) {
    const int H = 64 + i, W = 64 + i;
    const int Nn = 4, C = 128;
    const int P = Nn * H * W;
    const int gx = (P + 15) / 16;

    // conv1(3x3) + BN + ReLU -> h
    gemm_conv<0, 9><<<gx, 256, 0, stream>>>(xin, nullptr,
        w1 + (size_t)i * 147456, b1 + i * 128,
        g1 + i * 128, be1 + i * 128, m1 + i * 128, v1 + i * 128,
        hb, Nn, C, C, H, W, C * 9, 1);
    // offset conv (18 ch, no BN/ReLU)
    gemm_conv<0, 9><<<gx, 256, 0, stream>>>(hb, nullptr,
        w_off + (size_t)i * 20736, b_off + i * 18,
        nullptr, nullptr, nullptr, nullptr,
        offb, Nn, C, 18, H, W, C * 9, 0);
    // mask conv (9 ch, sigmoid applied in im2col)
    gemm_conv<0, 9><<<gx, 256, 0, stream>>>(hb, nullptr,
        w_mod + (size_t)i * 10368, b_mod + i * 9,
        nullptr, nullptr, nullptr, nullptr,
        mskb, Nn, C, 9, H, W, C * 9, 0);
    // NCHW -> NHWC for channel-contiguous bilinear gathers
    {
      const size_t total = (size_t)Nn * C * H * W;
      nchw_to_nhwc<<<(unsigned)((total + 255) / 256), 256, 0, stream>>>(hb, htb, Nn, C, H, W);
    }
    // modulated deformable im2col -> bf16 [P][1152]
    deform_im2col<<<dim3(H * W, Nn), 128, 0, stream>>>(htb, offb, mskb, vb, Nn, H, W);
    // deformable conv GEMM + BN -> d  (async global->LDS B staging)
    gemm_conv<1, 1><<<gx, 256, 0, stream>>>(nullptr, vb,
        w_def + (size_t)i * 147456, b_def + i * 128,
        g2 + i * 128, be2 + i * 128, m2 + i * 128, v2 + i * 128,
        db, Nn, 1152, C, H, W, 1152, 0);
    // 1x1 downsample conv + BN -> res
    gemm_conv<0, 1><<<gx, 256, 0, stream>>>(xin, nullptr,
        w_ds + (size_t)i * 16384, b_ds + i * 128,
        g3 + i * 128, be3 + i * 128, m3 + i * 128, v3 + i * 128,
        resb, Nn, C, C, H, W, C, 0);
    // relu(d + res) + stride-1 2x2 maxpool (pad 1) -> next x / output
    {
      float* outp = (i == 0) ? xb : (float*)d_out;
      const size_t total = (size_t)Nn * C * (H + 1) * (W + 1);
      add_relu_pool<<<(unsigned)((total + 255) / 256), 256, 0, stream>>>(db, resb, outp, Nn, C, H, W);
    }
    xin = xb;
  }
}
